// LatentActionMapper_86062554677977
// MI455X (gfx1250) — matrix operand order repaired and checked by reading, not compiled, not run on hardware
//
#include <hip/hip_runtime.h>
#include <hip/hip_bf16.h>

typedef __bf16 bf16;
typedef __attribute__((ext_vector_type(16))) __bf16 v16bf;
typedef __attribute__((ext_vector_type(8)))  __bf16 v8bf;
typedef __attribute__((ext_vector_type(8)))  float   v8f;

#define NTOK 32896   /* 8*16*257, divisible by 128 */
#define SEQ  257
#define BTF  128     /* B*T */
#define DM   512

#define F_BIAS   1
#define F_GELU   2
#define F_RES    4
#define F_STOREF 8
#define F_STOREB 16

__device__ __forceinline__ v16bf mk16(v8bf lo, v8bf hi) {
  v16bf r;
#pragma unroll
  for (int i = 0; i < 8; ++i) { r[i] = lo[i]; r[i + 8] = hi[i]; }
  return r;
}

__device__ __forceinline__ v8bf zero8() {
  v8bf z;
#pragma unroll
  for (int i = 0; i < 8; ++i) z[i] = (bf16)0.0f;
  return z;
}

__device__ __forceinline__ v8f wmma_bf16(v16bf a, v16bf b, v8f c) {
  return __builtin_amdgcn_wmma_f32_16x16x32_bf16(false, a, false, b, (short)0, c,
                                                 false, false);
}

__device__ __forceinline__ float gelu_tanh(float x) {
  float x3 = x * x * x;
  return 0.5f * x * (1.0f + tanhf(0.7978845608028654f * (x + 0.044715f * x3)));
}

// CDNA5 async copy: global -> LDS, 16 bytes per lane, tracked by ASYNCcnt.
// INST_OFFSET is added to BOTH the LDS and the global address (ISA 08, §4.4).
__device__ __forceinline__ void async_b128(unsigned lds_off, const void* gaddr,
                                           int ioff) {
  if (ioff == 0)
    asm volatile("global_load_async_to_lds_b128 %0, %1, off"
                 :: "v"(lds_off), "v"(gaddr) : "memory");
  else
    asm volatile("global_load_async_to_lds_b128 %0, %1, off offset:16"
                 :: "v"(lds_off), "v"(gaddr) : "memory");
}
// async loads complete in order -> waiting for <=4 outstanding guarantees the
// previous tile's 4 transfers have landed while the next tile is in flight.
__device__ __forceinline__ void wait_async_le(int n) {
  if (n == 0) asm volatile("s_wait_asynccnt 0x0" ::: "memory");
  else        asm volatile("s_wait_asynccnt 0x4" ::: "memory");
}

// ---------------------------------------------------------------- weights f32 -> bf16 transposed [N][K]
__global__ void k_convT(const float* __restrict__ src, bf16* __restrict__ dst,
                        int K, int N) {
  size_t i = (size_t)blockIdx.x * 256 + threadIdx.x;
  size_t total = (size_t)K * N;
  if (i >= total) return;
  int k = (int)(i / N), n = (int)(i % N);
  dst[(size_t)n * K + k] = (bf16)src[i];
}

// ---------------------------------------------------------------- patchify + action token -> X bf16 [NTOK][192]
__global__ void k_patchify(const float* __restrict__ videos,
                           const float* __restrict__ action_token,
                           bf16* __restrict__ X) {
  size_t i = (size_t)blockIdx.x * 256 + threadIdx.x;
  const size_t total = (size_t)NTOK * 192;
  if (i >= total) return;
  int col = (int)(i % 192);
  size_t tokn = i / 192;
  int s = (int)(tokn % SEQ);
  size_t bt = tokn / SEQ;  // b*16 + t
  float v;
  if (s == 0) {
    v = action_token[col];
  } else {
    int p = s - 1;
    int py = p >> 4, px = p & 15;
    int ch = col % 3;
    int pix = col / 3;
    int ph = pix >> 3, pw = pix & 7;
    int hh = py * 8 + ph, ww = px * 8 + pw;
    v = videos[(((bt)*128 + hh) * 128 + ww) * 3 + ch];
  }
  X[i] = (bf16)v;
}

// ---------------------------------------------------------------- layernorm over D=512, one wave per token -> bf16
__global__ __launch_bounds__(256) void k_layernorm(const float* __restrict__ x,
                                                   const float* __restrict__ g,
                                                   const float* __restrict__ b,
                                                   bf16* __restrict__ y, int ntok) {
  int wave = threadIdx.x >> 5;
  int lane = threadIdx.x & 31;
  int tok = blockIdx.x * 8 + wave;
  if (tok >= ntok) return;
  const float* row = x + (size_t)tok * DM;
  float s = 0.f, ss = 0.f;
  float v[16];
#pragma unroll
  for (int i = 0; i < 16; ++i) {
    float t = row[lane + 32 * i];
    v[i] = t; s += t; ss += t * t;
  }
#pragma unroll
  for (int off = 16; off; off >>= 1) {
    s += __shfl_xor(s, off, 32);
    ss += __shfl_xor(ss, off, 32);
  }
  float mean = s * (1.0f / DM);
  float var = ss * (1.0f / DM) - mean * mean;
  float rstd = rsqrtf(var + 1e-5f);
  bf16* yr = y + (size_t)tok * DM;
#pragma unroll
  for (int i = 0; i < 16; ++i) {
    int d = lane + 32 * i;
    yr[d] = (bf16)((v[i] - mean) * rstd * g[d] + b[d]);
  }
}

// ---------------------------------------------------------------- final LN on action tokens of frames 1..15 -> zin[120][512]
__global__ __launch_bounds__(32) void k_final_ln(const float* __restrict__ tok,
                                                 const float* __restrict__ g,
                                                 const float* __restrict__ b,
                                                 bf16* __restrict__ zin) {
  int r = blockIdx.x;            // 0..119
  int bb = r / 15, t = r % 15 + 1;
  size_t tidx = ((size_t)bb * 16 + t) * SEQ;  // s == 0
  int lane = threadIdx.x;
  const float* row = tok + tidx * DM;
  float s = 0.f, ss = 0.f;
  float v[16];
#pragma unroll
  for (int i = 0; i < 16; ++i) {
    float x = row[lane + 32 * i];
    v[i] = x; s += x; ss += x * x;
  }
#pragma unroll
  for (int off = 16; off; off >>= 1) {
    s += __shfl_xor(s, off, 32);
    ss += __shfl_xor(ss, off, 32);
  }
  float mean = s * (1.0f / DM);
  float var = ss * (1.0f / DM) - mean * mean;
  float rstd = rsqrtf(var + 1e-5f);
  bf16* yr = zin + (size_t)r * DM;
#pragma unroll
  for (int i = 0; i < 16; ++i) {
    int d = lane + 32 * i;
    yr[d] = (bf16)((v[i] - mean) * rstd * g[d] + b[d]);
  }
}

// ---------------------------------------------------------------- big WMMA GEMM: 128x128 block, 32x64 per wave (8 WMMAs/K-step)
// CONTRACT: M % 128 == 0, N % 128 == 0, K % 32 == 0.
// Double-buffered LDS, staged via CDNA5 global_load_async_to_lds_b128; each
// wave issues 4 async instrs per tile, pipelined with s_wait_asynccnt 4.
__global__ __launch_bounds__(256) void k_gemm_big(const bf16* __restrict__ A,
                                                  const bf16* __restrict__ Wt,
                                                  const float* __restrict__ bias,
                                                  float* __restrict__ outF,
                                                  bf16* __restrict__ outB,
                                                  int N, int K, int flags) {
  __shared__ bf16 lA[2][128][40];
  __shared__ bf16 lB[2][128][40];
  int tid = threadIdx.x;
  int lane = tid & 31, wave = tid >> 5;
  int bm = blockIdx.x * 128, bn = blockIdx.y * 128;
  int wm = (wave & 3) * 32;    // 0,32,64,96
  int wn = (wave >> 2) * 64;   // 0,64

  v8f acc[2][4];
#pragma unroll
  for (int i = 0; i < 2; ++i)
#pragma unroll
    for (int j = 0; j < 4; ++j)
#pragma unroll
      for (int r = 0; r < 8; ++r) acc[i][j][r] = 0.f;

  int srow = tid >> 1;          // 0..127
  int scol = (tid & 1) * 16;    // halves: 0 or 16

  const bf16* gA = A + (size_t)(bm + srow) * K + scol;
  const bf16* gB = Wt + (size_t)(bn + srow) * K + scol;
  unsigned la[2] = { (unsigned)(uintptr_t)&lA[0][srow][scol],
                     (unsigned)(uintptr_t)&lA[1][srow][scol] };
  unsigned lb[2] = { (unsigned)(uintptr_t)&lB[0][srow][scol],
                     (unsigned)(uintptr_t)&lB[1][srow][scol] };

  int koff  = (lane >> 4) * 8;
  int koff2 = (lane >> 4) * 16;
  int nl = lane & 15;
  int nt = K >> 5;

  // preload tile 0
  async_b128(la[0], gA, 0);
  async_b128(la[0], gA, 16);
  async_b128(lb[0], gB, 0);
  async_b128(lb[0], gB, 16);

  for (int t = 0; t < nt; ++t) {
    int cur = t & 1;
    if (t + 1 < nt) {
      int nxt = (t + 1) & 1;
      const bf16* pa = gA + (size_t)(t + 1) * 32;
      const bf16* pb = gB + (size_t)(t + 1) * 32;
      async_b128(la[nxt], pa, 0);
      async_b128(la[nxt], pa, 16);
      async_b128(lb[nxt], pb, 0);
      async_b128(lb[nxt], pb, 16);
      wait_async_le(4);
    } else {
      wait_async_le(0);
    }
    __syncthreads();

    v16bf af[2];
#pragma unroll
    for (int i = 0; i < 2; ++i) {
      int row = wm + i * 16 + nl;
      af[i] = mk16(*(const v8bf*)&lA[cur][row][koff],
                   *(const v8bf*)&lA[cur][row][16 + koff]);
    }
    v16bf bfg[4];
#pragma unroll
    for (int j = 0; j < 4; ++j) {
      int col = wn + j * 16 + nl;
      bfg[j] = mk16(*(const v8bf*)&lB[cur][col][koff],
                    *(const v8bf*)&lB[cur][col][16 + koff]);
    }
#pragma unroll
    for (int i = 0; i < 2; ++i)
#pragma unroll
      for (int j = 0; j < 4; ++j) acc[i][j] = wmma_bf16(af[i], bfg[j], acc[i][j]);
    __syncthreads();
  }

#pragma unroll
  for (int i = 0; i < 2; ++i) {
#pragma unroll
    for (int j = 0; j < 4; ++j) {
#pragma unroll
      for (int r = 0; r < 8; ++r) {
        int row = bm + wm + i * 16 + r + ((lane >> 4) << 3);
        int col = bn + wn + j * 16 + nl;
        float v = acc[i][j][r];
        if (flags & F_BIAS) v += bias[col];
        if (flags & F_GELU) v = gelu_tanh(v);
        size_t idx = (size_t)row * N + col;
        if (flags & F_RES) outF[idx] += v;
        if (flags & F_STOREF) outF[idx] = v;
        if (flags & F_STOREB) outB[idx] = (bf16)v;
      }
    }
  }
}

// ---------------------------------------------------------------- small generic WMMA GEMM (64x64 block) for the 120x64 tail
__global__ __launch_bounds__(256) void k_gemm(const bf16* __restrict__ A,
                                              const bf16* __restrict__ Wt,
                                              const float* __restrict__ bias,
                                              float* __restrict__ outF,
                                              bf16* __restrict__ outB,
                                              int M, int N, int K, int flags) {
  __shared__ bf16 lA[64][40];
  __shared__ bf16 lB[64][40];
  int tid = threadIdx.x;
  int lane = tid & 31, wave = tid >> 5;
  int bm = blockIdx.x * 64, bn = blockIdx.y * 64;
  int wm = (wave >> 1) * 16;
  int wn = (wave & 1) * 32;
  v8f acc0 = {}; v8f acc1 = {};

  int arow = tid >> 2, acol = (tid & 3) * 8;

  for (int k0 = 0; k0 < K; k0 += 32) {
    {
      int gr = bm + arow;
      v8bf av;
      if (gr < M) av = *(const v8bf*)(A + (size_t)gr * K + k0 + acol);
      else av = zero8();
      *(v8bf*)&lA[arow][acol] = av;
    }
    {
      v8bf wv = *(const v8bf*)(Wt + (size_t)(bn + arow) * K + k0 + acol);
      *(v8bf*)&lB[arow][acol] = wv;
    }
    __syncthreads();

    int row = wm + (lane & 15);
    int koff = (lane >> 4) * 8;
    v16bf af = mk16(*(const v8bf*)&lA[row][koff],
                    *(const v8bf*)&lA[row][16 + koff]);
    int koff2 = (lane >> 4) * 16;
    int c0 = wn + (lane & 15);
    v16bf b0 = mk16(*(const v8bf*)&lB[c0][koff],
                    *(const v8bf*)&lB[c0][16 + koff]);
    v16bf b1 = mk16(*(const v8bf*)&lB[c0 + 16][koff],
                    *(const v8bf*)&lB[c0 + 16][16 + koff]);
    acc0 = wmma_bf16(af, b0, acc0);
    acc1 = wmma_bf16(af, b1, acc1);
    __syncthreads();
  }

#pragma unroll
  for (int f = 0; f < 2; ++f) {
#pragma unroll
    for (int r = 0; r < 8; ++r) {
      int row = bm + wm + r + ((lane >> 4) << 3);
      int col = bn + wn + f * 16 + (lane & 15);
      if (row < M) {
        float v = (f == 0) ? acc0[r] : acc1[r];
        if (flags & F_BIAS) v += bias[col];
        if (flags & F_GELU) v = gelu_tanh(v);
        size_t idx = (size_t)row * N + col;
        if (flags & F_RES) outF[idx] += v;
        if (flags & F_STOREF) outF[idx] = v;
        if (flags & F_STOREB) outB[idx] = (bf16)v;
      }
    }
  }
}

// ---------------------------------------------------------------- spatial flash attention
// 4 waves per block, each wave owns one 16-query tile of the same (bt, head);
// V tile is staged into shared LDS cooperatively once per key-block.
__global__ __launch_bounds__(128) void k_attn_spatial(const bf16* __restrict__ qkv,
                                                      bf16* __restrict__ outp) {
  const int D3 = 1536;
  int lane = threadIdx.x & 31;
  int wv = threadIdx.x >> 5;
  int h = blockIdx.y, bt = blockIdx.z;
  int qt = blockIdx.x * 4 + wv;      // 0..19
  bool active = qt < 17;
  if (qt > 16) qt = 16;
  const bf16* base = qkv + (size_t)bt * SEQ * D3;
  int q0 = qt * 16;

  int koff  = (lane >> 4) * 8;
  int koff2 = (lane >> 4) * 16;
  int nloc  = lane & 15;

  v16bf aq[2];
  {
    int qs = q0 + nloc; if (qs > SEQ - 1) qs = SEQ - 1;
    const bf16* qrow = base + (size_t)qs * D3 + h * 64;
#pragma unroll
    for (int cc = 0; cc < 2; ++cc) {
      aq[cc] = mk16(*(const v8bf*)(qrow + cc * 32 + koff),
                    *(const v8bf*)(qrow + cc * 32 + 16 + koff));
    }
  }

  float mrun[8], lrun[8];
  v8f o[4];
#pragma unroll
  for (int r = 0; r < 8; ++r) { mrun[r] = -1e30f; lrun[r] = 0.f; }
#pragma unroll
  for (int dt = 0; dt < 4; ++dt)
#pragma unroll
    for (int r = 0; r < 8; ++r) o[dt][r] = 0.f;

  __shared__ bf16 lP[4][16][40];   // per-wave P tiles
  __shared__ bf16 lV[64][40];      // shared transposed V tile

  for (int kb = 0; kb < 9; ++kb) {
    int kbase = kb * 32;
    // --- S = Q K^T (two 16-key subtiles), K read straight from global ---
    v8f sa[2];
#pragma unroll
    for (int sub = 0; sub < 2; ++sub) {
      int ks = kbase + sub * 16 + nloc; if (ks > SEQ - 1) ks = SEQ - 1;
      const bf16* krow = base + (size_t)ks * D3 + DM + h * 64;
      v8f c = {};
#pragma unroll
      for (int cc = 0; cc < 2; ++cc) {
        v16bf bk = mk16(*(const v8bf*)(krow + cc * 32 + koff),
                        *(const v8bf*)(krow + cc * 32 + 16 + koff));
        c = wmma_bf16(aq[cc], bk, c);
      }
      sa[sub] = c;
    }
    // --- online softmax update ---
    bool valid0 = (kbase + nloc) < SEQ;
    bool valid1 = (kbase + 16 + nloc) < SEQ;
#pragma unroll
    for (int r = 0; r < 8; ++r) {
      float s0 = valid0 ? sa[0][r] * 0.125f : -1e30f;
      float s1 = valid1 ? sa[1][r] * 0.125f : -1e30f;
      float t = fmaxf(s0, s1);
#pragma unroll
      for (int off = 1; off < 16; off <<= 1) t = fmaxf(t, __shfl_xor(t, off, 32));
      float nm = fmaxf(mrun[r], t);
      float corr = __expf(mrun[r] - nm);
      float e0 = valid0 ? __expf(s0 - nm) : 0.f;
      float e1 = valid1 ? __expf(s1 - nm) : 0.f;
      float rs = e0 + e1;
#pragma unroll
      for (int off = 1; off < 16; off <<= 1) rs += __shfl_xor(rs, off, 32);
      lrun[r] = lrun[r] * corr + rs;
      mrun[r] = nm;
#pragma unroll
      for (int dt = 0; dt < 4; ++dt) o[dt][r] *= corr;
      int m = r + ((lane >> 4) << 3);
      lP[wv][m][nloc]      = (bf16)e0;
      lP[wv][m][16 + nloc] = (bf16)e1;
    }
    // --- stage V tile transposed into shared LDS: lV[d][key] (all 128 thr) ---
    {
      int kv = kbase + (threadIdx.x >> 2); if (kv > SEQ - 1) kv = SEQ - 1;
      int dseg = (threadIdx.x & 3) * 16;
      const bf16* vrow = base + (size_t)kv * D3 + 2 * DM + h * 64 + dseg;
      v8bf v0 = *(const v8bf*)(vrow);
      v8bf v1 = *(const v8bf*)(vrow + 8);
      int keyloc = threadIdx.x >> 2;
#pragma unroll
      for (int j = 0; j < 8; ++j) {
        lV[dseg + j][keyloc] = v0[j];
        lV[dseg + 8 + j][keyloc] = v1[j];
      }
    }
    __syncthreads();
    // --- O += P V ---
    v16bf ap = mk16(*(const v8bf*)&lP[wv][nloc][koff],
                    *(const v8bf*)&lP[wv][nloc][16 + koff]);
#pragma unroll
    for (int dt = 0; dt < 4; ++dt) {
      int cold = dt * 16 + nloc;
      v16bf bv = mk16(*(const v8bf*)&lV[cold][koff],
                      *(const v8bf*)&lV[cold][16 + koff]);
      o[dt] = wmma_bf16(ap, bv, o[dt]);
    }
    __syncthreads();
  }

#pragma unroll
  for (int dt = 0; dt < 4; ++dt) {
#pragma unroll
    for (int r = 0; r < 8; ++r) {
      int srow = q0 + r + ((lane >> 4) << 3);
      if (active && srow < SEQ) {
        float vv = o[dt][r] / lrun[r];
        outp[((size_t)bt * SEQ + srow) * DM + h * 64 + dt * 16 + nloc] = (bf16)vv;
      }
    }
  }
}

// ---------------------------------------------------------------- temporal causal attention (T=16, tiny -> VALU)
__global__ __launch_bounds__(128) void k_attn_temporal(const bf16* __restrict__ qkv,
                                                       bf16* __restrict__ outp) {
  const int D3 = 1536;
  int bs = blockIdx.x;
  int b = bs / SEQ, s = bs % SEQ;
  int tid = threadIdx.x;
  int h = tid >> 4, q = tid & 15;
  size_t tokbase = ((size_t)b * 16) * SEQ + s;

  float Q[64];
  {
    const bf16* qp = qkv + (tokbase + (size_t)q * SEQ) * D3 + h * 64;
#pragma unroll
    for (int d = 0; d < 64; ++d) Q[d] = (float)qp[d];
  }
  float p[16];
  float mx = -1e30f;
  for (int j = 0; j <= q; ++j) {
    const bf16* kp = qkv + (tokbase + (size_t)j * SEQ) * D3 + DM + h * 64;
    float acc = 0.f;
#pragma unroll
    for (int d = 0; d < 64; ++d) acc += Q[d] * (float)kp[d];
    acc *= 0.125f;
    p[j] = acc;
    mx = fmaxf(mx, acc);
  }
  float sum = 0.f;
  for (int j = 0; j <= q; ++j) { p[j] = __expf(p[j] - mx); sum += p[j]; }
  float inv = 1.0f / sum;
  bf16* op = outp + (tokbase + (size_t)q * SEQ) * DM + h * 64;
  for (int d = 0; d < 64; ++d) {
    float a = 0.f;
    for (int j = 0; j <= q; ++j)
      a += p[j] * (float)qkv[(tokbase + (size_t)j * SEQ) * D3 + 2 * DM + h * 64 + d];
    op[d] = (bf16)(a * inv);
  }
}

// ---------------------------------------------------------------- VQ argmin + action projection -> d_out[120][16]
__global__ __launch_bounds__(256) void k_vq(const float* __restrict__ z,
                                            const float* __restrict__ cb,
                                            const float* __restrict__ aw,
                                            float* __restrict__ outp) {
  __shared__ float zs[64];
  __shared__ float bd[256];
  __shared__ int bi[256];
  int r = blockIdx.x, tid = threadIdx.x;
  if (tid < 64) zs[tid] = z[(size_t)r * 64 + tid];
  __syncthreads();
  float best = 3.4e38f; int bidx = 0;
  for (int c = tid; c < 4096; c += 256) {
    const float* cr = cb + (size_t)c * 64;
    float d = 0.f;
#pragma unroll
    for (int k = 0; k < 64; ++k) { float df = zs[k] - cr[k]; d += df * df; }
    if (d < best || (d == best && c < bidx)) { best = d; bidx = c; }
  }
  bd[tid] = best; bi[tid] = bidx;
  __syncthreads();
  for (int off = 128; off; off >>= 1) {
    if (tid < off) {
      float d2 = bd[tid + off]; int i2 = bi[tid + off];
      if (d2 < bd[tid] || (d2 == bd[tid] && i2 < bi[tid])) { bd[tid] = d2; bi[tid] = i2; }
    }
    __syncthreads();
  }
  int idx = bi[0];
  if (tid < 16) {
    const float* cr = cb + (size_t)idx * 64;
    float a = 0.f;
#pragma unroll
    for (int k = 0; k < 64; ++k) a += cr[k] * aw[k * 16 + tid];
    outp[(size_t)r * 16 + tid] = a;
  }
}

// ================================================================ host
extern "C" void kernel_launch(void* const* d_in, const int* in_sizes, int n_in,
                              void* d_out, int out_size, void* d_ws, size_t ws_size,
                              hipStream_t stream) {
  (void)in_sizes; (void)n_in; (void)out_size; (void)ws_size;
  const float* videos       = (const float*)d_in[0];
  const float* patch_w      = (const float*)d_in[1];
  const float* patch_b      = (const float*)d_in[2];
  const float* action_token = (const float*)d_in[3];
  const float* ln1_g = (const float*)d_in[4];
  const float* ln1_b = (const float*)d_in[5];
  const float* wqkv_s = (const float*)d_in[6];
  const float* wo_s   = (const float*)d_in[7];
  const float* ln2_g = (const float*)d_in[8];
  const float* ln2_b = (const float*)d_in[9];
  const float* wqkv_t = (const float*)d_in[10];
  const float* wo_t   = (const float*)d_in[11];
  const float* ln3_g = (const float*)d_in[12];
  const float* ln3_b = (const float*)d_in[13];
  const float* mlp_w1 = (const float*)d_in[14];
  const float* mlp_b1 = (const float*)d_in[15];
  const float* mlp_w2 = (const float*)d_in[16];
  const float* mlp_b2 = (const float*)d_in[17];
  const float* lnf_g = (const float*)d_in[18];
  const float* lnf_b = (const float*)d_in[19];
  const float* out_w = (const float*)d_in[20];
  const float* codebook = (const float*)d_in[21];
  const float* action_w = (const float*)d_in[22];

  char* ws = (char*)d_ws;
  size_t off = 0;
  auto alloc = [&](size_t bytes) -> char* {
    char* p = ws + off;
    off += (bytes + 255) & ~(size_t)255;
    return p;
  };
  float* tok  = (float*)alloc((size_t)NTOK * DM * 4);
  bf16* bufA  = (bf16*)alloc((size_t)NTOK * DM * 2);
  bf16* bufB  = (bf16*)alloc((size_t)NTOK * 2048 * 2);
  bf16* Xp    = (bf16*)alloc((size_t)NTOK * 192 * 2);
  bf16* patchT = (bf16*)alloc((size_t)512 * 192 * 2);
  bf16* wqkvsT = (bf16*)alloc((size_t)4 * 1536 * 512 * 2);
  bf16* wosT   = (bf16*)alloc((size_t)4 * 512 * 512 * 2);
  bf16* wqkvtT = (bf16*)alloc((size_t)4 * 1536 * 512 * 2);
  bf16* wotT   = (bf16*)alloc((size_t)4 * 512 * 512 * 2);
  bf16* mlp1T  = (bf16*)alloc((size_t)4 * 2048 * 512 * 2);
  bf16* mlp2T  = (bf16*)alloc((size_t)4 * 512 * 2048 * 2);
  bf16* outT   = (bf16*)alloc((size_t)64 * 512 * 2);
  bf16* zin    = (bf16*)alloc((size_t)120 * 512 * 2);
  float* zf    = (float*)alloc((size_t)120 * 64 * 4);

  auto convT = [&](const float* src, bf16* dst, int K, int N) {
    size_t n = (size_t)K * N;
    k_convT<<<(unsigned)((n + 255) / 256), 256, 0, stream>>>(src, dst, K, N);
  };
  auto gemmBig = [&](const bf16* A, const bf16* Wt, const float* bias, float* oF,
                     bf16* oB, int M, int N, int K, int flags) {
    dim3 g(M / 128, N / 128);
    k_gemm_big<<<g, 256, 0, stream>>>(A, Wt, bias, oF, oB, N, K, flags);
  };

  // weight conversion (f32 -> bf16, transposed [N][K])
  convT(patch_w, patchT, 192, 512);
  for (int i = 0; i < 4; ++i) {
    convT(wqkv_s + (size_t)i * 512 * 1536, wqkvsT + (size_t)i * 1536 * 512, 512, 1536);
    convT(wo_s   + (size_t)i * 512 * 512,  wosT   + (size_t)i * 512 * 512,  512, 512);
    convT(wqkv_t + (size_t)i * 512 * 1536, wqkvtT + (size_t)i * 1536 * 512, 512, 1536);
    convT(wo_t   + (size_t)i * 512 * 512,  wotT   + (size_t)i * 512 * 512,  512, 512);
    convT(mlp_w1 + (size_t)i * 512 * 2048, mlp1T  + (size_t)i * 2048 * 512, 512, 2048);
    convT(mlp_w2 + (size_t)i * 2048 * 512, mlp2T  + (size_t)i * 512 * 2048, 2048, 512);
  }
  convT(out_w, outT, 512, 64);

  // patchify + embed
  {
    size_t n = (size_t)NTOK * 192;
    k_patchify<<<(unsigned)((n + 255) / 256), 256, 0, stream>>>(videos, action_token, Xp);
  }
  gemmBig(Xp, patchT, patch_b, tok, nullptr, NTOK, 512, 192, F_BIAS | F_STOREF);

  const unsigned lnGrid = (NTOK + 7) / 8;
  for (int i = 0; i < 4; ++i) {
    // --- spatial attention block ---
    k_layernorm<<<lnGrid, 256, 0, stream>>>(tok, ln1_g + i * 512, ln1_b + i * 512, bufA, NTOK);
    gemmBig(bufA, wqkvsT + (size_t)i * 1536 * 512, nullptr, nullptr, bufB,
            NTOK, 1536, 512, F_STOREB);
    k_attn_spatial<<<dim3(5, 8, BTF), 128, 0, stream>>>(bufB, bufA);
    gemmBig(bufA, wosT + (size_t)i * 512 * 512, nullptr, tok, nullptr,
            NTOK, 512, 512, F_RES);
    // --- temporal causal attention block ---
    k_layernorm<<<lnGrid, 256, 0, stream>>>(tok, ln2_g + i * 512, ln2_b + i * 512, bufA, NTOK);
    gemmBig(bufA, wqkvtT + (size_t)i * 1536 * 512, nullptr, nullptr, bufB,
            NTOK, 1536, 512, F_STOREB);
    k_attn_temporal<<<8 * SEQ, 128, 0, stream>>>(bufB, bufA);
    gemmBig(bufA, wotT + (size_t)i * 512 * 512, nullptr, tok, nullptr,
            NTOK, 512, 512, F_RES);
    // --- MLP block ---
    k_layernorm<<<lnGrid, 256, 0, stream>>>(tok, ln3_g + i * 512, ln3_b + i * 512, bufA, NTOK);
    gemmBig(bufA, mlp1T + (size_t)i * 2048 * 512, mlp_b1 + i * 2048, nullptr, bufB,
            NTOK, 2048, 512, F_BIAS | F_GELU | F_STOREB);
    gemmBig(bufB, mlp2T + (size_t)i * 512 * 2048, mlp_b2 + i * 512, tok, nullptr,
            NTOK, 512, 2048, F_BIAS | F_RES);
  }

  // final LN on selected action tokens, latent projection, VQ, action head
  k_final_ln<<<120, 32, 0, stream>>>(tok, lnf_g, lnf_b, zin);
  {
    dim3 g((120 + 63) / 64, 64 / 64);
    k_gemm<<<g, 256, 0, stream>>>(zin, outT, nullptr, zf, nullptr, 120, 64, 512,
                                  F_STOREF);
  }
  k_vq<<<120, 256, 0, stream>>>(zf, codebook, action_w, (float*)d_out);
}
